// DiTBlock_27281632264398
// MI455X (gfx1250) — matrix operand order repaired and checked by reading, not compiled, not run on hardware
//
#include <hip/hip_runtime.h>
#include <stdint.h>

typedef __bf16 bf16;
typedef __attribute__((ext_vector_type(16))) __bf16 v16bf;
typedef __attribute__((ext_vector_type(8)))  __bf16 v8bf;
typedef __attribute__((ext_vector_type(8)))  float  v8f;

#define D      1024
#define SIXD   6144
#define LSEQ   1024
#define NHEAD  16
#define HD     64
#define MLP_H  4096
#define ROWS   4096   // B*L

// ---------------- WMMA helpers ----------------

__device__ inline v8f wmma_bf16(v16bf a, v16bf b, v8f c) {
    return __builtin_amdgcn_wmma_f32_16x16x32_bf16(false, a, false, b, (short)0, c, false, false);
}

// A-fragment (16x32, MxK): lane=row, k chunks at +0 and +16 (each 8 contiguous bf16)
__device__ inline v16bf ld_a(const bf16* p) {
    v8bf lo = *(const v8bf*)(p);
    v8bf hi = *(const v8bf*)(p + 16);
    v16bf r;
#pragma unroll
    for (int i = 0; i < 8; ++i) { r[i] = lo[i]; r[i + 8] = hi[i]; }
    return r;
}

// B-fragment (32x16, KxN): lane=N-col, k rows at +0..7 and +16..23 (stride ld)
__device__ inline v16bf ld_b(const bf16* p, int ld) {
    v16bf r;
#pragma unroll
    for (int i = 0; i < 8; ++i) { r[i] = p[i * ld]; r[i + 8] = p[(i + 16) * ld]; }
    return r;
}

// Wait on outstanding async (global<->LDS) copies: ASYNCcnt <= N
template <int N>
__device__ inline void async_wait() {
#if __has_builtin(__builtin_amdgcn_s_wait_asynccnt)
    __builtin_amdgcn_s_wait_asynccnt(N);
#else
    if (N == 0) asm volatile("s_wait_asynccnt 0x0" ::: "memory");
    else        asm volatile("s_wait_asynccnt 0x1" ::: "memory");
#endif
}

// Per-lane 16B async copy global -> LDS (VDST = LDS byte addr, VADDR = 64b global addr)
__device__ inline void async_copy_b128(const bf16* gaddr, const bf16* lds_ptr) {
    unsigned ldsa = (unsigned)(size_t)lds_ptr;   // addrspace(3) offset = low 32 bits
    asm volatile("global_load_async_to_lds_b128 %0, %1, off"
                 :: "v"(ldsa), "v"(gaddr) : "memory");
}

__device__ inline float gelu_tanh(float x) {
    float x3 = x * x * x;
    return 0.5f * x * (1.0f + tanhf(0.7978845608028654f * (x + 0.044715f * x3)));
}

// ---------------- elementwise kernels ----------------

__global__ __launch_bounds__(256) void silu_to_bf16_kernel(const float* __restrict__ c,
                                                           bf16* __restrict__ out, int n) {
    int i = blockIdx.x * 256 + threadIdx.x;
    if (i < n) {
        float v = c[i];
        out[i] = (bf16)(v / (1.0f + __expf(-v)));
    }
}

__global__ __launch_bounds__(256) void f32_to_bf16_kernel(const float* __restrict__ in,
                                                          bf16* __restrict__ out, int n) {
    int i = blockIdx.x * 256 + threadIdx.x;
    if (i < n) out[i] = (bf16)in[i];
}

// LayerNorm (no affine) + adaLN modulate -> bf16. One block per row.
__global__ __launch_bounds__(256) void ln_modulate_kernel(const float* __restrict__ x,
                                                          const float* __restrict__ mod,
                                                          bf16* __restrict__ h,
                                                          int shift_off, int scale_off) {
    int row = blockIdx.x;
    int t = threadIdx.x;
    const float* xr = x + (size_t)row * D;
    float4 v = ((const float4*)xr)[t];
    float s  = v.x + v.y + v.z + v.w;
    float s2 = v.x * v.x + v.y * v.y + v.z * v.z + v.w * v.w;
#pragma unroll
    for (int m = 16; m >= 1; m >>= 1) {
        s  += __shfl_xor(s, m, 32);
        s2 += __shfl_xor(s2, m, 32);
    }
    __shared__ float rs[8], rs2[8];
    if ((t & 31) == 0) { rs[t >> 5] = s; rs2[t >> 5] = s2; }
    __syncthreads();
    float sum = 0.f, sum2 = 0.f;
#pragma unroll
    for (int i = 0; i < 8; ++i) { sum += rs[i]; sum2 += rs2[i]; }
    float mu   = sum * (1.0f / D);
    float var  = sum2 * (1.0f / D) - mu * mu;
    float rstd = rsqrtf(var + 1e-6f);
    const float* mrow = mod + (size_t)row * SIXD;
    int col = t * 4;
    float xv[4] = {v.x, v.y, v.z, v.w};
#pragma unroll
    for (int i = 0; i < 4; ++i) {
        float sc = mrow[scale_off + col + i];
        float sh = mrow[shift_off + col + i];
        h[(size_t)row * D + col + i] = (bf16)((xv[i] - mu) * rstd * (1.0f + sc) + sh);
    }
}

// qkv fp32 [B,L,3,H,HD] -> q [B,H,L,HD], kt [B,H,HD,L], v [B,H,L,HD]  (bf16)
__global__ __launch_bounds__(256) void repack_qkv_kernel(const float* __restrict__ qkv,
                                                         bf16* __restrict__ q,
                                                         bf16* __restrict__ kt,
                                                         bf16* __restrict__ v) {
    int i = blockIdx.x * 256 + threadIdx.x;   // [b][h][l][hd] flat
    int hd = i & 63;
    int l  = (i >> 6) & 1023;
    int h  = (i >> 16) & 15;
    int b  = i >> 20;
    size_t src = ((size_t)(b * LSEQ + l)) * (3 * D) + h * HD + hd;
    q[i] = (bf16)qkv[src];
    bf16 kv = (bf16)qkv[src + D];
    v[i] = (bf16)qkv[src + 2 * D];
    kt[((size_t)(b * NHEAD + h) * HD + hd) * LSEQ + l] = kv;
}

// ---------------- GEMM: C[M,N] = A[M,K](bf16) x B[K,N](bf16) + bias, fused epilogue ----
// Block = 8 waves, tile 128M x 64N. The 32x64 bf16 B tile is staged in LDS once per
// block via async global->LDS copies (ASYNCcnt), double-buffered across k-steps.
// MODE 0: out fp32 = acc + bias
// MODE 1: out bf16 = gelu(acc + bias)
// MODE 2: out fp32 = res + gate[row*gstride + col] * (acc + bias)
template <int MODE>
__global__ __launch_bounds__(256) void gemm_bf16_kernel(const bf16* __restrict__ A,
                                                        const bf16* __restrict__ Bm,
                                                        const float* __restrict__ bias,
                                                        const float* __restrict__ res,
                                                        const float* __restrict__ gate,
                                                        int gstride, void* outp,
                                                        int N, int Kd) {
    __shared__ __align__(16) bf16 btile[2][32 * 64];   // double-buffered 4KB B tile

    int t    = threadIdx.x;
    int wave = t >> 5;
    int lane = t & 31;
    int lrow = lane & 15;
    int half = lane >> 4;
    int m0 = blockIdx.x * 128 + wave * 16;
    int n0 = blockIdx.y * 64;

    const bf16* Ap = A + (size_t)(m0 + lrow) * Kd + half * 8;   // A-frag base
    int krow = lrow + half * 16;                                 // B-frag K row

    // async-copy assignment: 256 threads x 16B = 4KB tile (32 rows x 64 bf16)
    int cr = t >> 3;            // tile row 0..31
    int cc = (t & 7) * 8;       // element col within row (8 bf16 = 16B)
    const bf16* gB = Bm + n0 + cc;

    // prologue: stage k-step 0 into buffer 0
    async_copy_b128(gB + (size_t)cr * N, &btile[0][cr * 64 + cc]);

    v8f acc0 = {}, acc1 = {}, acc2 = {}, acc3 = {};
    int nk = Kd >> 5;
    for (int i = 0; i < nk; ++i) {
        int cur = i & 1;
        if (i + 1 < nk) {
            async_copy_b128(gB + (size_t)((i + 1) * 32 + cr) * N,
                            &btile[cur ^ 1][cr * 64 + cc]);
            async_wait<1>();            // buffer `cur` complete (in-order), next in flight
        } else {
            async_wait<0>();
        }
        __syncthreads();                // make buffer `cur` visible to all waves

        v16bf a = ld_a(Ap + i * 32);
        const bf16* bp = &btile[cur][half * 8 * 64 + lrow];
        v16bf b0 = ld_b(bp, 64);
        v16bf b1 = ld_b(bp + 16, 64);
        v16bf b2 = ld_b(bp + 32, 64);
        v16bf b3 = ld_b(bp + 48, 64);
        acc0 = wmma_bf16(a, b0, acc0);
        acc1 = wmma_bf16(a, b1, acc1);
        acc2 = wmma_bf16(a, b2, acc2);
        acc3 = wmma_bf16(a, b3, acc3);

        __syncthreads();                // all waves done reading before re-issue into cur
    }

    v8f accs[4] = {acc0, acc1, acc2, acc3};
#pragma unroll
    for (int j = 0; j < 4; ++j) {
#pragma unroll
        for (int e = 0; e < 8; ++e) {
            int row = m0 + e + 8 * half;
            int col = n0 + j * 16 + lrow;
            float vv = accs[j][e] + bias[col];
            size_t idx = (size_t)row * N + col;
            if (MODE == 0) {
                ((float*)outp)[idx] = vv;
            } else if (MODE == 1) {
                ((bf16*)outp)[idx] = (bf16)gelu_tanh(vv);
            } else {
                ((float*)outp)[idx] = res[idx] + gate[(size_t)row * gstride + col] * vv;
            }
        }
    }
}

// ---------------- Flash attention: one wave per (b, h, 16-query tile) ----------------
__global__ __launch_bounds__(32) void flash_attn_kernel(const bf16* __restrict__ q,
                                                        const bf16* __restrict__ kt,
                                                        const bf16* __restrict__ v,
                                                        const unsigned char* __restrict__ mask,
                                                        bf16* __restrict__ attn_out) {
    int qb = blockIdx.x * 16;
    int h  = blockIdx.y;
    int b  = blockIdx.z;
    int lane = threadIdx.x;
    int lrow = lane & 15;
    int half = lane >> 4;

    const bf16* Q  = q  + ((size_t)(b * NHEAD + h) * LSEQ) * HD;
    const bf16* KT = kt + ((size_t)(b * NHEAD + h) * HD) * LSEQ;
    const bf16* V  = v  + ((size_t)(b * NHEAD + h) * LSEQ) * HD;
    const unsigned char* Mrow = mask + b * LSEQ;

    // Q fragments for the two 32-wide k-steps over HD=64
    const bf16* Qp = Q + (size_t)(qb + lrow) * HD + half * 8;
    v16bf qa0 = ld_a(Qp);
    v16bf qa1 = ld_a(Qp + 32);

    float m_i[8], l_i[8];
#pragma unroll
    for (int e = 0; e < 8; ++e) { m_i[e] = -1e30f; l_i[e] = 0.f; }
    v8f acc[4] = {};

    __shared__ __align__(16) bf16 pbuf[16 * 32];
    int khd = half * 8;                // B-frag K base row

    for (int kb = 0; kb < LSEQ; kb += 32) {
        // S = Q Kt : two 16x16 tiles (keys kb..+15, kb+16..+31)
        const bf16* kp0 = KT + (size_t)khd * LSEQ + kb + lrow;          // hd = khd
        const bf16* kp1 = KT + (size_t)(khd + 32) * LSEQ + kb + lrow;   // hd = khd+32
        v8f s0 = {}, s1 = {};
        s0 = wmma_bf16(qa0, ld_b(kp0, LSEQ), s0);
        s0 = wmma_bf16(qa1, ld_b(kp1, LSEQ), s0);
        s1 = wmma_bf16(qa0, ld_b(kp0 + 16, LSEQ), s1);
        s1 = wmma_bf16(qa1, ld_b(kp1 + 16, LSEQ), s1);

        bool mk0 = Mrow[kb + lrow] != 0;
        bool mk1 = Mrow[kb + 16 + lrow] != 0;

        float alpha[8];
#pragma unroll
        for (int e = 0; e < 8; ++e) {
            float v0 = mk0 ? -1e30f : s0[e] * 0.125f;   // HD^-0.5 = 1/8
            float v1 = mk1 ? -1e30f : s1[e] * 0.125f;
            float r = fmaxf(v0, v1);
#pragma unroll
            for (int mm = 8; mm >= 1; mm >>= 1) r = fmaxf(r, __shfl_xor(r, mm, 32));
            float mn = fmaxf(m_i[e], r);
            float al = __expf(m_i[e] - mn);
            float p0 = __expf(v0 - mn);
            float p1 = __expf(v1 - mn);
            float ps = p0 + p1;
#pragma unroll
            for (int mm = 8; mm >= 1; mm >>= 1) ps += __shfl_xor(ps, mm, 32);
            l_i[e] = l_i[e] * al + ps;
            m_i[e] = mn;
            alpha[e] = al;
            int row = e + 8 * half;
            pbuf[row * 32 + lrow]      = (bf16)p0;
            pbuf[row * 32 + 16 + lrow] = (bf16)p1;
        }
        __syncthreads();

#pragma unroll
        for (int j = 0; j < 4; ++j)
#pragma unroll
            for (int e = 0; e < 8; ++e) acc[j][e] *= alpha[e];

        // P fragment (16x32) from LDS, then O += P V
        v16bf pa = ld_a(&pbuf[lrow * 32 + half * 8]);
        const bf16* vp = V + (size_t)(kb + khd) * HD + lrow;
#pragma unroll
        for (int j = 0; j < 4; ++j)
            acc[j] = wmma_bf16(pa, ld_b(vp + j * 16, HD), acc[j]);
        __syncthreads();
    }

    float inv[8];
#pragma unroll
    for (int e = 0; e < 8; ++e) inv[e] = (l_i[e] > 0.f) ? 1.0f / l_i[e] : 0.f;

#pragma unroll
    for (int j = 0; j < 4; ++j)
#pragma unroll
        for (int e = 0; e < 8; ++e) {
            int row = qb + e + 8 * half;
            int col = h * HD + j * 16 + lrow;
            attn_out[(size_t)(b * LSEQ + row) * D + col] = (bf16)(acc[j][e] * inv[e]);
        }
}

// ---------------- host launch ----------------

extern "C" void kernel_launch(void* const* d_in, const int* in_sizes, int n_in,
                              void* d_out, int out_size, void* d_ws, size_t ws_size,
                              hipStream_t stream) {
    (void)in_sizes; (void)n_in; (void)out_size; (void)ws_size;
    const float* x      = (const float*)d_in[0];
    const float* c      = (const float*)d_in[1];
    const unsigned char* mask = (const unsigned char*)d_in[2];
    const float* w_ada  = (const float*)d_in[3];
    const float* b_ada  = (const float*)d_in[4];
    const float* w_qkv  = (const float*)d_in[5];
    const float* b_qkv  = (const float*)d_in[6];
    const float* w_proj = (const float*)d_in[7];
    const float* b_proj = (const float*)d_in[8];
    const float* w_fc1  = (const float*)d_in[9];
    const float* b_fc1  = (const float*)d_in[10];
    const float* w_fc2  = (const float*)d_in[11];
    const float* b_fc2  = (const float*)d_in[12];
    float* out = (float*)d_out;

    const size_t MB = 1ull << 20;
    char* w = (char*)d_ws;
    float* mod   = (float*)(w + 0);             // 96 MB  [4096,6144] fp32
    bf16*  hbuf  = (bf16*)(w + 96  * MB);       //  8 MB  [4096,1024] bf16
    float* x1    = (float*)(w + 104 * MB);      // 16 MB  [4096,1024] fp32
    char*  P     = w + 120 * MB;                // reuse pool
    bf16*  csilu = (bf16*)(P);                  //  8 MB
    bf16*  wada  = (bf16*)(P + 8 * MB);         // 12 MB
    bf16*  wqkv  = (bf16*)(P);                  //  6 MB   (after mod GEMM)
    float* qkv   = (float*)(P + 8 * MB);        // 48 MB
    bf16*  qb    = (bf16*)(P + 56 * MB);        //  8 MB
    bf16*  ktb   = (bf16*)(P + 64 * MB);        //  8 MB
    bf16*  vb    = (bf16*)(P + 72 * MB);        //  8 MB
    bf16*  attn  = (bf16*)(P);                  //  8 MB   (after repack)
    bf16*  wproj = (bf16*)(P + 8 * MB);         //  2 MB
    bf16*  wfc1  = (bf16*)(P + 12 * MB);        //  8 MB
    bf16*  hmlp  = (bf16*)(P + 20 * MB);        // 32 MB
    bf16*  wfc2  = (bf16*)(P + 52 * MB);        //  8 MB

    // 1. silu(c) -> bf16
    silu_to_bf16_kernel<<<ROWS * D / 256, 256, 0, stream>>>(c, csilu, ROWS * D);
    // 2. mod = silu(c) @ w_ada + b_ada
    f32_to_bf16_kernel<<<D * SIXD / 256, 256, 0, stream>>>(w_ada, wada, D * SIXD);
    gemm_bf16_kernel<0><<<dim3(ROWS / 128, SIXD / 64), 256, 0, stream>>>(
        csilu, wada, b_ada, nullptr, nullptr, 0, mod, SIXD, D);
    // 3. h = modulate(LN(x), shift_msa, scale_msa)
    ln_modulate_kernel<<<ROWS, 256, 0, stream>>>(x, mod, hbuf, 0, D);
    // 4. qkv = h @ w_qkv + b_qkv
    f32_to_bf16_kernel<<<D * 3 * D / 256, 256, 0, stream>>>(w_qkv, wqkv, D * 3 * D);
    gemm_bf16_kernel<0><<<dim3(ROWS / 128, 3 * D / 64), 256, 0, stream>>>(
        hbuf, wqkv, b_qkv, nullptr, nullptr, 0, qkv, 3 * D, D);
    // 5. repack to q/kt/v
    repack_qkv_kernel<<<ROWS * D / 256, 256, 0, stream>>>(qkv, qb, ktb, vb);
    // 6. flash attention -> attn bf16 [B,L,D]
    flash_attn_kernel<<<dim3(LSEQ / 16, NHEAD, 4), 32, 0, stream>>>(qb, ktb, vb, mask, attn);
    // 7. x1 = x + gate_msa * (attn @ w_proj + b_proj)
    f32_to_bf16_kernel<<<D * D / 256, 256, 0, stream>>>(w_proj, wproj, D * D);
    gemm_bf16_kernel<2><<<dim3(ROWS / 128, D / 64), 256, 0, stream>>>(
        attn, wproj, b_proj, x, mod + 2 * D, SIXD, x1, D, D);
    // 8. h = modulate(LN(x1), shift_mlp, scale_mlp)
    ln_modulate_kernel<<<ROWS, 256, 0, stream>>>(x1, mod, hbuf, 3 * D, 4 * D);
    // 9. hmlp = gelu(h @ w_fc1 + b_fc1) -> bf16
    f32_to_bf16_kernel<<<D * MLP_H / 256, 256, 0, stream>>>(w_fc1, wfc1, D * MLP_H);
    gemm_bf16_kernel<1><<<dim3(ROWS / 128, MLP_H / 64), 256, 0, stream>>>(
        hbuf, wfc1, b_fc1, nullptr, nullptr, 0, hmlp, MLP_H, D);
    // 10. out = x1 + gate_mlp * (hmlp @ w_fc2 + b_fc2)
    f32_to_bf16_kernel<<<MLP_H * D / 256, 256, 0, stream>>>(w_fc2, wfc2, MLP_H * D);
    gemm_bf16_kernel<2><<<dim3(ROWS / 128, D / 64), 256, 0, stream>>>(
        hmlp, wfc2, b_fc2, x1, mod + 5 * D, SIXD, out, D, MLP_H);
}
